// LncTracker_21517786153308
// MI455X (gfx1250) — matrix operand order repaired and checked by reading, not compiled, not run on hardware
//
#include <hip/hip_runtime.h>
#include <math.h>

typedef _Float16 half_t;
typedef __attribute__((ext_vector_type(16))) _Float16 v16h;
typedef __attribute__((ext_vector_type(8)))  float    v8f;

// Problem constants (match the reference)
constexpr int Bz = 4, Lz = 1024, Vz = 64, Dz = 256, Hz = 4, DKz = 64,
              NLz = 4, Mz = 20, FFz = 1024, HDz = Hz * DKz;   // HDz = 256

// ---------------------------------------------------------------------------
// WMMA helpers (CDNA5 gfx1250, wave32). D = A(16x32 f16) * B(32x16 f16) + C(f32)
// ---------------------------------------------------------------------------
static __device__ __forceinline__ v8f wmma_f16(v16h a, v16h b, v8f c) {
  // 8 args: (neg_a, A, neg_b, B, c_mod, C, reuse_a, reuse_b)
  return __builtin_amdgcn_wmma_f32_16x16x32_f16(false, a, false, b, (short)0, c,
                                                false, false);
}

// A-fragment, 16x32 f16, row-major A with leading dim lda.
// Layout: lanes 0-15 row m, K in k0..k0+15;
//         lanes 16-31 row m, K in k0+16..k0+31 (matches load_fragB).
static __device__ __forceinline__ v16h load_fragA(const half_t* __restrict__ A,
                                                  int lda, int row, int k0,
                                                  int lane) {
  const int m  = row + (lane & 15);
  const int kb = k0 + ((lane >> 4) << 4);           // +16 for upper half-wave
  const half_t* p = A + (long)m * lda + kb;
  union { v16h v; uint4 u[2]; } r;
  r.u[0] = *(const uint4*)(p);                      // K = kb .. kb+7
  r.u[1] = *(const uint4*)(p + 8);                  // K = kb+8 .. kb+15
  return r.v;
}

// B-fragment, 32x16 f16, stored transposed: Bt is [N,K] row-major (ldb = K dim).
// lanes 0-15 col n with K k0..k0+15; lanes 16-31 col n with K k0+16..k0+31.
static __device__ __forceinline__ v16h load_fragB(const half_t* __restrict__ Bt,
                                                  int ldb, int col, int k0,
                                                  int lane) {
  const int n  = col + (lane & 15);
  const int kb = k0 + ((lane >> 4) << 4);           // +16 for upper half-wave
  return *(const v16h*)(Bt + (long)n * ldb + kb);
}

// ---------------------------------------------------------------------------
// Small utility kernels
// ---------------------------------------------------------------------------
__global__ void zero_f32_kernel(float* __restrict__ p, int n) {
  int i = blockIdx.x * 256 + threadIdx.x;
  if (i < n) p[i] = 0.0f;
}

__global__ void f32_to_f16_kernel(const float* __restrict__ in,
                                  half_t* __restrict__ out, int n) {
  int i = blockIdx.x * 256 + threadIdx.x;
  if (i < n) out[i] = (half_t)in[i];
}

// W [K,N] f32 row-major  ->  Wt [N,K] f16 row-major
__global__ void transpose_w_kernel(const float* __restrict__ W,
                                   half_t* __restrict__ Wt, int K, int N) {
  int i = blockIdx.x * 256 + threadIdx.x;
  if (i < K * N) {
    int k = i / N, n = i - k * N;
    Wt[(long)n * K + k] = (half_t)W[i];
  }
}

// v [B*L, H*DK] f16 -> vT [(b*H+h)*DK + dk][L] f16
__global__ void transpose_v_kernel(const half_t* __restrict__ v,
                                   half_t* __restrict__ vT) {
  int idx = blockIdx.x * 256 + threadIdx.x;         // B*L*H*DK = 1<<20
  if (idx >= Bz * Lz * HDz) return;
  int dk = idx & (DKz - 1);
  int h  = (idx >> 6) & (Hz - 1);
  int l  = (idx >> 8) & (Lz - 1);
  int b  = idx >> 18;
  long src = ((long)(b * Lz + l)) * HDz + h * DKz + dk;
  long dst = (((long)(b * Hz + h) * DKz + dk)) * Lz + l;
  vT[dst] = v[src];
}

// ---------------------------------------------------------------------------
// Generic batched f16 WMMA GEMM:  C[z] = A[z] (MxK) * Bt[z]^T (KxN) + bias
// Block = 256 threads = 8 waves (4 x 2), block tile 128 x 64, wave tile 32 x 32.
// grid = (N/64, M/128, nz).  z -> (zb = z/Hd, zh = z%Hd) for composite strides.
// act: 0 = none, 1 = exact GELU. Cf (f32) and/or Ch (f16) outputs, both optional.
// ---------------------------------------------------------------------------
__global__ void gemm_f16_kernel(const half_t* __restrict__ A, int lda, long sAb, long sAh,
                                const half_t* __restrict__ Bt, int ldb, long sBb, long sBh,
                                const float* __restrict__ bias,
                                float* __restrict__ Cf, half_t* __restrict__ Ch,
                                int ldc, long sCb, long sCh,
                                int K, int act, int Hd) {
  const int z  = blockIdx.z;
  const int zb = z / Hd, zh = z % Hd;
  A  += zb * sAb + zh * sAh;
  Bt += zb * sBb + zh * sBh;
  const long coff = zb * sCb + zh * sCh;

  const int lane = threadIdx.x & 31;
  const int wave = threadIdx.x >> 5;
  const int row0 = blockIdx.y * 128 + (wave >> 1) * 32;
  const int col0 = blockIdx.x * 64  + (wave & 1)  * 32;

  v8f acc00 = {}, acc01 = {}, acc10 = {}, acc11 = {};
  for (int k0 = 0; k0 < K; k0 += 32) {
    if (k0 + 32 < K) {  // prefetch next K tile of A rows this lane touches
      __builtin_prefetch(A + (long)(row0 + (lane & 15)) * lda + k0 + 32, 0, 1);
      __builtin_prefetch(Bt + (long)(col0 + (lane & 15)) * ldb + k0 + 32, 0, 1);
    }
    v16h a0 = load_fragA(A, lda, row0,      k0, lane);
    v16h a1 = load_fragA(A, lda, row0 + 16, k0, lane);
    v16h b0 = load_fragB(Bt, ldb, col0,      k0, lane);
    v16h b1 = load_fragB(Bt, ldb, col0 + 16, k0, lane);
    acc00 = wmma_f16(a0, b0, acc00);
    acc01 = wmma_f16(a0, b1, acc01);
    acc10 = wmma_f16(a1, b0, acc10);
    acc11 = wmma_f16(a1, b1, acc11);
  }

  const int rup = (lane >> 4) << 3;   // +8 rows for lanes 16-31
  const int nc  = lane & 15;
  auto emit = [&](float v, int row, int col) {
    if (bias) v += bias[col];
    if (act)  v = 0.5f * v * (1.0f + erff(v * 0.70710678118654752f));
    long o = coff + (long)row * ldc + col;
    if (Cf) Cf[o] = v;
    if (Ch) Ch[o] = (half_t)v;
  };
#pragma unroll
  for (int r = 0; r < 8; ++r) {
    emit(acc00[r], row0 + rup + r,      col0 + nc);
    emit(acc01[r], row0 + rup + r,      col0 + 16 + nc);
    emit(acc10[r], row0 + 16 + rup + r, col0 + nc);
    emit(acc11[r], row0 + 16 + rup + r, col0 + 16 + nc);
  }
}

// ---------------------------------------------------------------------------
// Attention scores:  S[b,h,i,j] = (q_i . k_j) / sqrt(DK) * bias_k + bias_b
// Fused relative-position log-bucket bias in the WMMA epilogue.
// grid = (L/64, L/128, B*H)
// ---------------------------------------------------------------------------
__global__ void scores_kernel(const half_t* __restrict__ q,
                              const half_t* __restrict__ k,
                              const float* __restrict__ relk,   // [2M+1, H]
                              const float* __restrict__ relb,   // [2M+1, H]
                              float* __restrict__ scores) {
  const int z = blockIdx.z;
  const int b = z >> 2, h = z & 3;
  const half_t* A  = q + ((long)b * Lz) * HDz + h * DKz;
  const half_t* Bt = k + ((long)b * Lz) * HDz + h * DKz;
  float* S = scores + (long)z * Lz * Lz;

  const int lane = threadIdx.x & 31;
  const int wave = threadIdx.x >> 5;
  const int row0 = blockIdx.y * 128 + (wave >> 1) * 32;   // i
  const int col0 = blockIdx.x * 64  + (wave & 1)  * 32;   // j

  v8f acc00 = {}, acc01 = {}, acc10 = {}, acc11 = {};
#pragma unroll
  for (int k0 = 0; k0 < DKz; k0 += 32) {
    v16h a0 = load_fragA(A, HDz, row0,      k0, lane);
    v16h a1 = load_fragA(A, HDz, row0 + 16, k0, lane);
    v16h b0 = load_fragB(Bt, HDz, col0,      k0, lane);
    v16h b1 = load_fragB(Bt, HDz, col0 + 16, k0, lane);
    acc00 = wmma_f16(a0, b0, acc00);
    acc01 = wmma_f16(a0, b1, acc01);
    acc10 = wmma_f16(a1, b0, acc10);
    acc11 = wmma_f16(a1, b1, acc11);
  }

  const int rup = (lane >> 4) << 3;
  const int nc  = lane & 15;
  auto emit = [&](float v, int i, int j) {
    float s = v * 0.125f;                       // 1/sqrt(64)
    int d = i - j; if (d < 0) d = -d;
    float df = (float)d;
    if (d > Mz) df = (float)Mz + log2f(fmaxf((float)(d - Mz), 1.0f));
    if (df > (float)(2 * Mz)) df = (float)(2 * Mz);
    int idx = (int)df;                          // truncation, matches astype(int32)
    float bk = relk[idx * Hz + h];
    float bb = relb[idx * Hz + h];
    S[(long)i * Lz + j] = s * bk + bb;
  };
#pragma unroll
  for (int r = 0; r < 8; ++r) {
    emit(acc00[r], row0 + rup + r,      col0 + nc);
    emit(acc01[r], row0 + rup + r,      col0 + 16 + nc);
    emit(acc10[r], row0 + 16 + rup + r, col0 + nc);
    emit(acc11[r], row0 + 16 + rup + r, col0 + 16 + nc);
  }
}

// ---------------------------------------------------------------------------
// Softmax over j (L=1024) per row (b,h,i). Writes f16 probs and accumulates
// prob/NL into the avg-attn output region. grid = B*H*L, block = 256.
// ---------------------------------------------------------------------------
__global__ void softmax_kernel(const float* __restrict__ scores,
                               half_t* __restrict__ probs,
                               float* __restrict__ avg) {
  const long base = (long)blockIdx.x * Lz;
  const int t = threadIdx.x;
  __shared__ float red[256];

  float v[4];
  float mx = -3.4e38f;
#pragma unroll
  for (int s = 0; s < 4; ++s) {
    v[s] = scores[base + t + 256 * s];
    mx = fmaxf(mx, v[s]);
  }
  red[t] = mx; __syncthreads();
  for (int s = 128; s > 0; s >>= 1) {
    if (t < s) red[t] = fmaxf(red[t], red[t + s]);
    __syncthreads();
  }
  mx = red[0]; __syncthreads();

  float e[4], acc = 0.0f;
#pragma unroll
  for (int s = 0; s < 4; ++s) { e[s] = __expf(v[s] - mx); acc += e[s]; }
  red[t] = acc; __syncthreads();
  for (int s = 128; s > 0; s >>= 1) {
    if (t < s) red[t] += red[t + s];
    __syncthreads();
  }
  const float inv = 1.0f / red[0];
#pragma unroll
  for (int s = 0; s < 4; ++s) {
    float p = e[s] * inv;
    probs[base + t + 256 * s] = (half_t)p;
    avg[base + t + 256 * s] += p * (1.0f / (float)NLz);
  }
}

// ---------------------------------------------------------------------------
// LayerNorm over D=256. One block per row; writes f16 (and optional f32).
// ---------------------------------------------------------------------------
__global__ void layernorm_kernel(const float* __restrict__ X,
                                 const float* __restrict__ g,
                                 const float* __restrict__ bb,
                                 half_t* __restrict__ outh,
                                 float* __restrict__ outf) {
  const int row = blockIdx.x, t = threadIdx.x;
  const float x = X[(long)row * Dz + t];
  __shared__ float red[256];
  red[t] = x; __syncthreads();
  for (int s = 128; s > 0; s >>= 1) { if (t < s) red[t] += red[t + s]; __syncthreads(); }
  const float mu = red[0] * (1.0f / Dz); __syncthreads();
  const float dx = x - mu;
  red[t] = dx * dx; __syncthreads();
  for (int s = 128; s > 0; s >>= 1) { if (t < s) red[t] += red[t + s]; __syncthreads(); }
  const float var = red[0] * (1.0f / Dz);
  const float y = dx * rsqrtf(var + 1e-5f) * g[t] + bb[t];
  outh[(long)row * Dz + t] = (half_t)y;
  if (outf) outf[(long)row * Dz + t] = y;
}

// out[b*D+d] = mean over L of h[b,l,d].  grid = B*D/256, block = 256.
__global__ void mean_kernel(const float* __restrict__ h, float* __restrict__ out) {
  int idx = blockIdx.x * 256 + threadIdx.x;     // B*D = 1024
  if (idx >= Bz * Dz) return;
  int b = idx >> 8, d = idx & 255;
  float acc = 0.0f;
  for (int l = 0; l < Lz; ++l) acc += h[((long)(b * Lz + l)) * Dz + d];
  out[idx] = acc * (1.0f / Lz);
}

// ---------------------------------------------------------------------------
// Host orchestration
// ---------------------------------------------------------------------------
extern "C" void kernel_launch(void* const* d_in, const int* in_sizes, int n_in,
                              void* d_out, int out_size, void* d_ws, size_t ws_size,
                              hipStream_t stream) {
  (void)in_sizes; (void)n_in; (void)ws_size;
  const float* x     = (const float*)d_in[0];
  const float* emb   = (const float*)d_in[1];
  const float* Wq    = (const float*)d_in[2];
  const float* bq    = (const float*)d_in[3];
  const float* Wk    = (const float*)d_in[4];
  const float* bk    = (const float*)d_in[5];
  const float* Wv    = (const float*)d_in[6];
  const float* bv    = (const float*)d_in[7];
  const float* Wo    = (const float*)d_in[8];
  const float* bo    = (const float*)d_in[9];
  const float* rel_k = (const float*)d_in[10];
  const float* rel_b = (const float*)d_in[11];
  const float* ln1g  = (const float*)d_in[12];
  const float* ln1b  = (const float*)d_in[13];
  const float* W1    = (const float*)d_in[14];
  const float* b1    = (const float*)d_in[15];
  const float* W2    = (const float*)d_in[16];
  const float* b2    = (const float*)d_in[17];
  const float* ln2g  = (const float*)d_in[18];
  const float* ln2b  = (const float*)d_in[19];
  float* outp = (float*)d_out;

  // ---- workspace carve (256B aligned) ----
  size_t off = 0;
  auto take = [&](size_t bytes) -> void* {
    void* p = (char*)d_ws + off;
    off = (off + bytes + 255) & ~(size_t)255;
    return p;
  };
  const long BL = (long)Bz * Lz;                     // 4096
  half_t* x_h    = (half_t*)take(BL * Vz * 2);
  half_t* embT   = (half_t*)take((size_t)Dz * Vz * 2);
  half_t* WqT    = (half_t*)take((size_t)NLz * Dz * HDz * 2);
  half_t* WkT    = (half_t*)take((size_t)NLz * Dz * HDz * 2);
  half_t* WvT    = (half_t*)take((size_t)NLz * Dz * HDz * 2);
  half_t* WoT    = (half_t*)take((size_t)NLz * HDz * Dz * 2);
  half_t* W1T    = (half_t*)take((size_t)NLz * Dz * FFz * 2);
  half_t* W2T    = (half_t*)take((size_t)NLz * FFz * Dz * 2);
  half_t* hh     = (half_t*)take(BL * Dz * 2);
  half_t* q_h    = (half_t*)take(BL * HDz * 2);
  half_t* k_h    = (half_t*)take(BL * HDz * 2);
  half_t* v_h    = (half_t*)take(BL * HDz * 2);
  half_t* vT     = (half_t*)take(BL * HDz * 2);
  float*  scores = (float*) take((size_t)Bz * Hz * Lz * Lz * 4);   // 64 MB
  half_t* probs  = (half_t*)take((size_t)Bz * Hz * Lz * Lz * 2);   // 32 MB
  half_t* ao_h   = (half_t*)take(BL * HDz * 2);
  float*  y0_f   = (float*) take(BL * Dz * 4);
  half_t* y_h    = (half_t*)take(BL * Dz * 2);
  half_t* ff_h   = (half_t*)take(BL * FFz * 2);
  float*  y2_f   = (float*) take(BL * Dz * 4);
  float*  h_f    = (float*) take(BL * Dz * 4);

  float* avg_out = outp + Bz * Dz;                  // [B,H,L,L] region

  // ---- zero output (we accumulate avg-attn into it) ----
  zero_f32_kernel<<<(out_size + 255) / 256, 256, 0, stream>>>(outp, out_size);

  // ---- precision conversions / weight transposes ----
  f32_to_f16_kernel<<<(int)((BL * Vz + 255) / 256), 256, 0, stream>>>(x, x_h, (int)(BL * Vz));
  transpose_w_kernel<<<(Vz * Dz + 255) / 256, 256, 0, stream>>>(emb, embT, Vz, Dz);
  for (int l = 0; l < NLz; ++l) {
    transpose_w_kernel<<<(Dz * HDz + 255) / 256, 256, 0, stream>>>(Wq + (long)l * Dz * HDz, WqT + (long)l * HDz * Dz, Dz, HDz);
    transpose_w_kernel<<<(Dz * HDz + 255) / 256, 256, 0, stream>>>(Wk + (long)l * Dz * HDz, WkT + (long)l * HDz * Dz, Dz, HDz);
    transpose_w_kernel<<<(Dz * HDz + 255) / 256, 256, 0, stream>>>(Wv + (long)l * Dz * HDz, WvT + (long)l * HDz * Dz, Dz, HDz);
    transpose_w_kernel<<<(HDz * Dz + 255) / 256, 256, 0, stream>>>(Wo + (long)l * HDz * Dz, WoT + (long)l * Dz * HDz, HDz, Dz);
    transpose_w_kernel<<<(Dz * FFz + 255) / 256, 256, 0, stream>>>(W1 + (long)l * Dz * FFz, W1T + (long)l * FFz * Dz, Dz, FFz);
    transpose_w_kernel<<<(FFz * Dz + 255) / 256, 256, 0, stream>>>(W2 + (long)l * FFz * Dz, W2T + (long)l * Dz * FFz, FFz, Dz);
  }

  // ---- embedding: hh = f16( x @ emb ) ----
  gemm_f16_kernel<<<dim3(Dz / 64, (int)(BL / 128), 1), 256, 0, stream>>>(
      x_h, Vz, 0, 0, embT, Vz, 0, 0, nullptr,
      nullptr, hh, Dz, 0, 0, Vz, 0, 1);

  // ---- transformer layers ----
  for (int l = 0; l < NLz; ++l) {
    const half_t* WqT_l = WqT + (long)l * HDz * Dz;
    const half_t* WkT_l = WkT + (long)l * HDz * Dz;
    const half_t* WvT_l = WvT + (long)l * HDz * Dz;
    const half_t* WoT_l = WoT + (long)l * Dz * HDz;
    const half_t* W1T_l = W1T + (long)l * FFz * Dz;
    const half_t* W2T_l = W2T + (long)l * Dz * FFz;

    const dim3 gProj(HDz / 64, (int)(BL / 128), 1);
    gemm_f16_kernel<<<gProj, 256, 0, stream>>>(hh, Dz, 0, 0, WqT_l, Dz, 0, 0,
        bq + (long)l * HDz, nullptr, q_h, HDz, 0, 0, Dz, 0, 1);
    gemm_f16_kernel<<<gProj, 256, 0, stream>>>(hh, Dz, 0, 0, WkT_l, Dz, 0, 0,
        bk + (long)l * HDz, nullptr, k_h, HDz, 0, 0, Dz, 0, 1);
    gemm_f16_kernel<<<gProj, 256, 0, stream>>>(hh, Dz, 0, 0, WvT_l, Dz, 0, 0,
        bv + (long)l * HDz, nullptr, v_h, HDz, 0, 0, Dz, 0, 1);

    transpose_v_kernel<<<(int)((BL * HDz + 255) / 256), 256, 0, stream>>>(v_h, vT);

    scores_kernel<<<dim3(Lz / 64, Lz / 128, Bz * Hz), 256, 0, stream>>>(
        q_h, k_h,
        rel_k + (long)l * (2 * Mz + 1) * Hz,
        rel_b + (long)l * (2 * Mz + 1) * Hz, scores);

    softmax_kernel<<<Bz * Hz * Lz, 256, 0, stream>>>(scores, probs, avg_out);

    // attn @ v : per (b,h): [L,L] x [L,DK] -> ao_h[b*L + l, h*DK + dk]
    gemm_f16_kernel<<<dim3(DKz / 64, Lz / 128, Bz * Hz), 256, 0, stream>>>(
        probs, Lz, (long)Hz * Lz * Lz, (long)Lz * Lz,
        vT, Lz, (long)Hz * DKz * Lz, (long)DKz * Lz,
        nullptr, nullptr, ao_h, HDz, (long)Lz * HDz, (long)DKz,
        Lz, 0, Hz);

    // out projection -> f32 for LN1
    gemm_f16_kernel<<<dim3(Dz / 64, (int)(BL / 128), 1), 256, 0, stream>>>(
        ao_h, HDz, 0, 0, WoT_l, HDz, 0, 0, bo + (long)l * Dz,
        y0_f, nullptr, Dz, 0, 0, HDz, 0, 1);

    layernorm_kernel<<<(int)BL, 256, 0, stream>>>(
        y0_f, ln1g + (long)l * Dz, ln1b + (long)l * Dz, y_h, nullptr);

    // FFN with exact GELU
    gemm_f16_kernel<<<dim3(FFz / 64, (int)(BL / 128), 1), 256, 0, stream>>>(
        y_h, Dz, 0, 0, W1T_l, Dz, 0, 0, b1 + (long)l * FFz,
        nullptr, ff_h, FFz, 0, 0, Dz, 1, 1);
    gemm_f16_kernel<<<dim3(Dz / 64, (int)(BL / 128), 1), 256, 0, stream>>>(
        ff_h, FFz, 0, 0, W2T_l, FFz, 0, 0, b2 + (long)l * Dz,
        y2_f, nullptr, Dz, 0, 0, FFz, 0, 1);

    layernorm_kernel<<<(int)BL, 256, 0, stream>>>(
        y2_f, ln2g + (long)l * Dz, ln2b + (long)l * Dz, hh,
        (l == NLz - 1) ? h_f : nullptr);
  }

  // ---- final mean over L ----
  mean_kernel<<<(Bz * Dz + 255) / 256, 256, 0, stream>>>(h_f, outp);
}
